// negative_set_define_76647986364840
// MI455X (gfx1250) — matrix operand order repaired and checked by reading, not compiled, not run on hardware
//
#include <hip/hip_runtime.h>

typedef float  v4f    __attribute__((ext_vector_type(4)));
typedef float  v8f    __attribute__((ext_vector_type(8)));
typedef __bf16 bf16x4  __attribute__((ext_vector_type(4)));
typedef __bf16 bf16x8  __attribute__((ext_vector_type(8)));
typedef __bf16 bf16x16 __attribute__((ext_vector_type(16)));

#define N_ROWS 8192
#define DIM    1024
#define BM     128
#define BN     128
#define KB     64          // K panel per LDS stage (2 WMMA k-steps of 32)
#define LDSS   72          // bf16 units per LDS row: KB + 8 pad (16B-aligned, conflict-free)
#define COL_CHUNKS 8
#define TILES_PER_CHUNK 8  // (N_ROWS / COL_CHUNKS) / BN

// ---------------------------------------------------------------------------
// Kernel 1: n2[j] = ||negative[j]||^2 in full f32 ; one wave32 per row
// ---------------------------------------------------------------------------
__launch_bounds__(256)
__global__ void norms_kernel(const float* __restrict__ Bn, float* __restrict__ n2) {
    const int row  = blockIdx.x * 8 + (threadIdx.x >> 5);
    const int lane = threadIdx.x & 31;
    const float* p = Bn + (size_t)row * DIM;
    float s = 0.0f;
#pragma unroll
    for (int k = lane * 4; k < DIM; k += 32 * 4) {
        v4f v = *(const v4f*)(p + k);
        s += v.x * v.x + v.y * v.y + v.z * v.z + v.w * v.w;
    }
#pragma unroll
    for (int off = 16; off >= 1; off >>= 1) s += __shfl_xor(s, off, 32);
    if (lane == 0) n2[row] = s;
}

// ---------------------------------------------------------------------------
// Fragment gather helpers (bf16 WMMA 16x16x32 layouts, wave32)
//  A 16x32: lane(m=l16, half): elems[0..7]=K(8h..8h+7), elems[8..15]=K(16+8h..)
//  B 32x16: lane(n=l16, half): elems[0..7]=K(8h..8h+7), elems[8..15]=K(16+8h..)
// ---------------------------------------------------------------------------
__device__ __forceinline__ bf16x16 fragA(const __bf16* rowp, int half, int ks) {
    bf16x8 p0 = *(const bf16x8*)(rowp + ks + 8 * half);
    bf16x8 p1 = *(const bf16x8*)(rowp + ks + 16 + 8 * half);
    return __builtin_shufflevector(p0, p1, 0,1,2,3,4,5,6,7,8,9,10,11,12,13,14,15);
}
__device__ __forceinline__ bf16x16 fragB(const __bf16* rowp, int half, int ks) {
    bf16x8 p0 = *(const bf16x8*)(rowp + ks + 8 * half);
    bf16x8 p1 = *(const bf16x8*)(rowp + ks + 16 + 8 * half);
    return __builtin_shufflevector(p0, p1, 0,1,2,3,4,5,6,7,8,9,10,11,12,13,14,15);
}

// ---------------------------------------------------------------------------
// Kernel 2: split-bf16 WMMA GEMM (anchor x negative^T), 3 bf16 products per
// f32 product (hi*hi + hi*lo + lo*hi), f32 accumulate; fused per-row argmax
// of score = n2[j] - 2*dot. Partials per (row, column-chunk) to workspace.
// ---------------------------------------------------------------------------
__launch_bounds__(256)
__global__ void gemm_argmax_kernel(const float* __restrict__ A,
                                   const float* __restrict__ Bn,
                                   const float* __restrict__ n2,
                                   float* __restrict__ pvals,
                                   int*   __restrict__ pidx) {
    __shared__ __bf16 ldsAhi[BM * LDSS];
    __shared__ __bf16 ldsAlo[BM * LDSS];
    __shared__ __bf16 ldsBhi[BN * LDSS];
    __shared__ __bf16 ldsBlo[BN * LDSS];
    __shared__ float  redV[BM * 4];
    __shared__ int    redI[BM * 4];

    const int rb   = blockIdx.x;           // row block (64)
    const int cc   = blockIdx.y;           // column chunk (8)
    const int tid  = threadIdx.x;
    const int wave = tid >> 5;             // 0..7
    const int lane = tid & 31;
    const int half = lane >> 4;            // 0..1
    const int l16  = lane & 15;
    const int waveM = wave >> 2;           // 0..1  -> 64 rows each
    const int waveN = wave & 3;            // 0..3  -> 32 cols each
    const int rowBase = rb * BM;

    float bestV[4][8];
    int   bestI[4][8];
#pragma unroll
    for (int mt = 0; mt < 4; ++mt)
#pragma unroll
        for (int r = 0; r < 8; ++r) { bestV[mt][r] = -3.4e38f; bestI[mt][r] = 0; }

    for (int it = 0; it < TILES_PER_CHUNK; ++it) {
        const int colBase = cc * (BN * TILES_PER_CHUNK) + it * BN;

        v8f acc[4][2];
#pragma unroll
        for (int mt = 0; mt < 4; ++mt)
#pragma unroll
            for (int nt = 0; nt < 2; ++nt)
#pragma unroll
                for (int r = 0; r < 8; ++r) acc[mt][nt][r] = 0.0f;

        for (int kc = 0; kc < DIM; kc += KB) {
            __syncthreads();   // previous K-panel compute must finish
            // stage K panel: 128 rows x 64 f32 each for A and B, split to
            // bf16 hi/lo in registers, store 8B chunks to LDS.
#pragma unroll
            for (int i = 0; i < 8; ++i) {
                const int flat = tid + 256 * i;     // 0..2047
                const int row  = flat >> 4;         // 0..127
                const int seg  = flat & 15;         // 0..15 (4 floats each)
                const size_t offA = (size_t)(rowBase + row) * DIM + kc + seg * 4;
                const size_t offB = (size_t)(colBase + row) * DIM + kc + seg * 4;
                v4f va = *(const v4f*)(A + offA);
                v4f vb = *(const v4f*)(Bn + offB);
                if (kc + KB < DIM) {  // warm L2 for the next panel
                    __builtin_prefetch(A + offA + KB, 0, 1);
                    __builtin_prefetch(Bn + offB + KB, 0, 1);
                }
                bf16x4 ah, al, bh, bl;
#pragma unroll
                for (int e = 0; e < 4; ++e) {
                    float  xa = va[e];
                    __bf16 ha = (__bf16)xa;
                    ah[e] = ha; al[e] = (__bf16)(xa - (float)ha);
                    float  xb = vb[e];
                    __bf16 hb = (__bf16)xb;
                    bh[e] = hb; bl[e] = (__bf16)(xb - (float)hb);
                }
                *(bf16x4*)(ldsAhi + row * LDSS + seg * 4) = ah;
                *(bf16x4*)(ldsAlo + row * LDSS + seg * 4) = al;
                *(bf16x4*)(ldsBhi + row * LDSS + seg * 4) = bh;
                *(bf16x4*)(ldsBlo + row * LDSS + seg * 4) = bl;
            }
            __syncthreads();

#pragma unroll
            for (int ks = 0; ks < KB; ks += 32) {
                bf16x16 bHi[2], bLo[2];
#pragma unroll
                for (int nt = 0; nt < 2; ++nt) {
                    const int br = (waveN * 32 + nt * 16 + l16) * LDSS;
                    bHi[nt] = fragB(ldsBhi + br, half, ks);
                    bLo[nt] = fragB(ldsBlo + br, half, ks);
                }
#pragma unroll
                for (int mt = 0; mt < 4; ++mt) {
                    const int ar = (waveM * 64 + mt * 16 + l16) * LDSS;
                    bf16x16 aHi = fragA(ldsAhi + ar, half, ks);
                    bf16x16 aLo = fragA(ldsAlo + ar, half, ks);
#pragma unroll
                    for (int nt = 0; nt < 2; ++nt) {
                        acc[mt][nt] = __builtin_amdgcn_wmma_f32_16x16x32_bf16(
                            false, aHi, false, bLo[nt], (short)0, acc[mt][nt], false, false);
                        acc[mt][nt] = __builtin_amdgcn_wmma_f32_16x16x32_bf16(
                            false, aLo, false, bHi[nt], (short)0, acc[mt][nt], false, false);
                        acc[mt][nt] = __builtin_amdgcn_wmma_f32_16x16x32_bf16(
                            false, aHi, false, bHi[nt], (short)0, acc[mt][nt], false, false);
                    }
                }
            }
        }

        // fold this column tile into the running per-lane best
#pragma unroll
        for (int nt = 0; nt < 2; ++nt) {
            const int j  = colBase + waveN * 32 + nt * 16 + l16;
            const float nn = n2[j];
#pragma unroll
            for (int mt = 0; mt < 4; ++mt)
#pragma unroll
                for (int r = 0; r < 8; ++r) {
                    const float s = nn - 2.0f * acc[mt][nt][r];
                    if (s > bestV[mt][r] || (s == bestV[mt][r] && j < bestI[mt][r])) {
                        bestV[mt][r] = s; bestI[mt][r] = j;
                    }
                }
        }
    }

    // cross-lane argmax within each 16-lane half (C layout: lanes hold columns)
#pragma unroll
    for (int mt = 0; mt < 4; ++mt)
#pragma unroll
        for (int r = 0; r < 8; ++r) {
            float v = bestV[mt][r];
            int   ix = bestI[mt][r];
#pragma unroll
            for (int off = 8; off >= 1; off >>= 1) {
                const float ov = __shfl_xor(v, off, 32);
                const int   oi = __shfl_xor(ix, off, 32);
                if (ov > v || (ov == v && oi < ix)) { v = ov; ix = oi; }
            }
            if (l16 == 0) {
                const int row = waveM * 64 + mt * 16 + half * 8 + r;  // 0..127
                redV[row * 4 + waveN] = v;
                redI[row * 4 + waveN] = ix;
            }
        }
    __syncthreads();

    // cross-wave (4 column waves) reduce, write partial per (row, chunk)
    if (tid < BM) {
        float v = redV[tid * 4];
        int  ix = redI[tid * 4];
#pragma unroll
        for (int w = 1; w < 4; ++w) {
            const float ov = redV[tid * 4 + w];
            const int   oi = redI[tid * 4 + w];
            if (ov > v || (ov == v && oi < ix)) { v = ov; ix = oi; }
        }
        const int gr = rowBase + tid;
        pvals[gr * COL_CHUNKS + cc] = v;
        pidx [gr * COL_CHUNKS + cc] = ix;
    }
}

// ---------------------------------------------------------------------------
// Kernel 3: reduce the 8 chunk partials per row, gather negative[idx] row.
// ---------------------------------------------------------------------------
__launch_bounds__(256)
__global__ void gather_kernel(const float* __restrict__ Bn,
                              const float* __restrict__ pvals,
                              const int*   __restrict__ pidx,
                              float* __restrict__ out) {
    __shared__ int sIdx;
    const int row = blockIdx.x;
    const int tid = threadIdx.x;
    if (tid == 0) {
        float v = pvals[row * COL_CHUNKS];
        int  ix = pidx [row * COL_CHUNKS];
#pragma unroll
        for (int c = 1; c < COL_CHUNKS; ++c) {
            const float ov = pvals[row * COL_CHUNKS + c];
            const int   oi = pidx [row * COL_CHUNKS + c];
            if (ov > v || (ov == v && oi < ix)) { v = ov; ix = oi; }
        }
        sIdx = ix;
    }
    __syncthreads();
    const int ix = sIdx;
    v4f v = *(const v4f*)(Bn + (size_t)ix * DIM + tid * 4);
    *(v4f*)(out + (size_t)row * DIM + tid * 4) = v;
}

// ---------------------------------------------------------------------------
extern "C" void kernel_launch(void* const* d_in, const int* in_sizes, int n_in,
                              void* d_out, int out_size, void* d_ws, size_t ws_size,
                              hipStream_t stream) {
    (void)in_sizes; (void)n_in; (void)out_size; (void)ws_size;
    const float* anchor   = (const float*)d_in[0];
    const float* negative = (const float*)d_in[1];
    float* out = (float*)d_out;

    // workspace layout: n2[8192] | pvals[8192*8] | pidx[8192*8]  (~544 KB)
    float* n2    = (float*)d_ws;
    float* pvals = n2 + N_ROWS;
    int*   pidx  = (int*)(pvals + N_ROWS * COL_CHUNKS);

    norms_kernel<<<N_ROWS / 8, 256, 0, stream>>>(negative, n2);
    gemm_argmax_kernel<<<dim3(N_ROWS / BM, COL_CHUNKS), 256, 0, stream>>>(
        anchor, negative, n2, pvals, pidx);
    gather_kernel<<<N_ROWS, 256, 0, stream>>>(negative, pvals, pidx, out);
}
